// DeltaNetEnhancedAttention_23235773072087
// MI455X (gfx1250) — matrix operand order repaired and checked by reading, not compiled, not run on hardware
//
#include <hip/hip_runtime.h>
#include <hip/hip_bf16.h>

// ---------------- CDNA5 WMMA types ----------------
typedef __bf16 bf16_t;
typedef __attribute__((ext_vector_type(16))) __bf16 v16bf;
typedef __attribute__((ext_vector_type(8)))  float  v8f;

__device__ __forceinline__ v8f wmma_bf16(v16bf a, v16bf b, v8f c) {
  // D = A(16x32 bf16) * B(32x16 bf16) + C(16x16 f32)
  return __builtin_amdgcn_wmma_f32_16x16x32_bf16(
      false, a, false, b, (short)0, c, false, false);
}

__device__ __forceinline__ bf16_t f2bf(float f) {
  union { float f; unsigned u; } c; c.f = f;
  unsigned r = c.u + 0x7FFFu + ((c.u >> 16) & 1u);   // round-to-nearest-even
  union { unsigned short s; bf16_t b; } o; o.s = (unsigned short)(r >> 16);
  return o.b;
}

// 16 contiguous bf16 -> v16bf (two b128 loads)
__device__ __forceinline__ v16bf load_frag16(const bf16_t* p) {
  union { v16bf v; uint4 u[2]; } f;
  f.u[0] = *(const uint4*)p;
  f.u[1] = *(const uint4*)(p + 8);
  return f.v;
}
// A-fragment: elems 0..15 = K 2*kb0..2*kb0+15 (contiguous, matches B frags)
__device__ __forceinline__ v16bf load_fragA(const bf16_t* p, int kb0) {
  union { v16bf v; uint4 u[2]; } f;
  f.u[0] = *(const uint4*)(p + 2 * kb0);
  f.u[1] = *(const uint4*)(p + 2 * kb0 + 8);
  return f.v;
}

// ---------------- gfx1250 async global->LDS staging (16B per lane) ----------------
#if defined(__AMDGCN__) && defined(__has_builtin)
#  if __has_builtin(__builtin_amdgcn_global_load_async_to_lds_b128) && \
      __has_builtin(__builtin_amdgcn_s_wait_asynccnt)
#    define ASYNC_LDS 1
#  endif
#endif
#ifndef ASYNC_LDS
#  define ASYNC_LDS 0
#endif

typedef int v4i_t __attribute__((ext_vector_type(4)));
typedef __attribute__((address_space(1))) v4i_t g_v4i;   // global (printed __device__)
typedef __attribute__((address_space(3))) v4i_t l_v4i;   // LDS    (printed __shared__)

__device__ __forceinline__ void stage16(const bf16_t* g, bf16_t* l) {
#if ASYNC_LDS
  // generic LDS address: low 32 bits are the DS byte offset (aperture rules)
  __builtin_amdgcn_global_load_async_to_lds_b128(
      (g_v4i*)(size_t)g, (l_v4i*)(unsigned)(size_t)l, 0, 0);
#else
  *(uint4*)l = *(const uint4*)g;
#endif
}
__device__ __forceinline__ void stage_wait() {
#if ASYNC_LDS
  __builtin_amdgcn_s_wait_asynccnt(0);
#endif
}

// ---------------- Problem constants ----------------
#define BB 2
#define NN 3072
#define CC 1024
#define HH 16
#define DD 64
#define WINSZ 256
#define STRIDEW 192
#define NWIN 16
#define PP 5

// ---------------- f32 -> bf16 convert ----------------
__global__ void cvt_bf16_kernel(const float* __restrict__ src,
                                bf16_t* __restrict__ dst, size_t n) {
  size_t i = (size_t)blockIdx.x * blockDim.x + threadIdx.x;
  if (i < n) dst[i] = f2bf(src[i]);
}

// ---------------- Tiled bf16 WMMA GEMM: C = A(MxK) * B(KxN) (+bias) ----------------
// double-buffered LDS, software-pipelined staging
#define BM 128
#define BN 128
#define BK 32
#define BKP 40   // padded row: 80B stride -> 20-bank step, conflict-free b128 frags
__global__ __launch_bounds__(256) void gemm_bf16_kernel(
    const bf16_t* __restrict__ A, const bf16_t* __restrict__ Bm,
    float* __restrict__ C, const float* __restrict__ bias,
    int M, int Nn, int K) {
  __shared__ bf16_t As[2][BM][BKP];     // row-major MxK tiles
  __shared__ bf16_t Bs[2][BN][BKP];     // TRANSPOSED: Bs[buf][n][k]
  const int bm = blockIdx.y * BM, bn = blockIdx.x * BN;
  const int tid = threadIdx.x, lane = tid & 31, wv = tid >> 5;
  const int wr = wv >> 2, wc = wv & 3;        // 2x4 wave grid -> 64x32 per wave
  const int laneM = lane & 15, half = lane >> 4;
  const int kb0 = half ? 8 : 0, kbB = half ? 16 : 0;
  // per-thread staging coordinates (two 8-elem chunks each for A and B)
  const int i0 = tid * 16;
  const int ar0 = i0 >> 5, ac0 = i0 & 31;
  const int ar1 = (i0 + 8) >> 5, ac1 = (i0 + 8) & 31;
  const int br0 = i0 >> 7, bc0 = i0 & 127;
  const int br1 = (i0 + 8) >> 7, bc1 = (i0 + 8) & 127;

  const v8f vzero = {0.f, 0.f, 0.f, 0.f, 0.f, 0.f, 0.f, 0.f};
  v8f acc[4][2];
#pragma unroll
  for (int mt = 0; mt < 4; ++mt)
#pragma unroll
    for (int nt = 0; nt < 2; ++nt) acc[mt][nt] = vzero;

  // ---- prologue: stage tile 0 into buffer 0 ----
  {
    stage16(A + (size_t)(bm + ar0) * K + ac0, &As[0][ar0][ac0]);
    stage16(A + (size_t)(bm + ar1) * K + ac1, &As[0][ar1][ac1]);
    union { uint4 u; bf16_t e[8]; } t0, t1;
    t0.u = *(const uint4*)(Bm + (size_t)br0 * Nn + bn + bc0);
    t1.u = *(const uint4*)(Bm + (size_t)br1 * Nn + bn + bc1);
#pragma unroll
    for (int i = 0; i < 8; ++i) Bs[0][bc0 + i][br0] = t0.e[i];
#pragma unroll
    for (int i = 0; i < 8; ++i) Bs[0][bc1 + i][br1] = t1.e[i];
    stage_wait();
  }
  __syncthreads();

  for (int kt = 0; kt < K; kt += BK) {
    const int cur = (kt / BK) & 1, nxt = cur ^ 1;
    const bool hasNext = (kt + BK) < K;
    union { uint4 u; bf16_t e[8]; } tb0, tb1;
#if !ASYNC_LDS
    uint4 ta0, ta1;
#endif
    if (hasNext) {
      const bf16_t* sA0 = A + (size_t)(bm + ar0) * K + kt + BK + ac0;
      const bf16_t* sA1 = A + (size_t)(bm + ar1) * K + kt + BK + ac1;
      const bf16_t* sB0 = Bm + (size_t)(kt + BK + br0) * Nn + bn + bc0;
      const bf16_t* sB1 = Bm + (size_t)(kt + BK + br1) * Nn + bn + bc1;
#if ASYNC_LDS
      stage16(sA0, &As[nxt][ar0][ac0]);
      stage16(sA1, &As[nxt][ar1][ac1]);
#else
      ta0 = *(const uint4*)sA0;
      ta1 = *(const uint4*)sA1;
#endif
      tb0.u = *(const uint4*)sB0;
      tb1.u = *(const uint4*)sB1;
      __builtin_prefetch(sA0 + BK, 0, 1);                 // speculative, kt+2 tile
      __builtin_prefetch(sB0 + (size_t)BK * Nn, 0, 1);
    }

    // ---- compute from buffer `cur` ----
    v16bf afr[4], bfr[2];
#pragma unroll
    for (int mt = 0; mt < 4; ++mt)
      afr[mt] = load_fragA(&As[cur][wr * 64 + mt * 16 + laneM][0], kb0);
#pragma unroll
    for (int nt = 0; nt < 2; ++nt)
      bfr[nt] = load_frag16(&Bs[cur][wc * 32 + nt * 16 + laneM][kbB]);
#pragma unroll
    for (int mt = 0; mt < 4; ++mt)
#pragma unroll
      for (int nt = 0; nt < 2; ++nt)
        acc[mt][nt] = wmma_bf16(afr[mt], bfr[nt], acc[mt][nt]);

    // ---- finish staging into buffer `nxt` ----
    if (hasNext) {
#if !ASYNC_LDS
      *(uint4*)&As[nxt][ar0][ac0] = ta0;
      *(uint4*)&As[nxt][ar1][ac1] = ta1;
#endif
#pragma unroll
      for (int i = 0; i < 8; ++i) Bs[nxt][bc0 + i][br0] = tb0.e[i];
#pragma unroll
      for (int i = 0; i < 8; ++i) Bs[nxt][bc1 + i][br1] = tb1.e[i];
      stage_wait();
    }
    __syncthreads();
  }
  // epilogue (C layout: VGPR r -> M=r + 8*half)
#pragma unroll
  for (int mt = 0; mt < 4; ++mt)
#pragma unroll
    for (int nt = 0; nt < 2; ++nt) {
      int colg = bn + wc * 32 + nt * 16 + laneM;
      float bv = bias ? bias[colg] : 0.f;
#pragma unroll
      for (int r = 0; r < 8; ++r) {
        int row = bm + wr * 64 + mt * 16 + r + half * 8;
        C[(size_t)row * Nn + colg] = acc[mt][nt][r] + bv;
      }
    }
}

// ---------------- RMS-norm q,k + bf16 repack of q,k,v into (B,H,N,D) ----------------
__global__ __launch_bounds__(256) void qkv_post_kernel(
    const float* __restrict__ qkv, bf16_t* __restrict__ qnb,
    bf16_t* __restrict__ knb, bf16_t* __restrict__ vb) {
  int row = blockIdx.x * 8 + (threadIdx.x >> 5);  // (b*H+h)*N + n
  int lane = threadIdx.x & 31;
  int n = row % NN, bh = row / NN;
  int h = bh % HH, b = bh / HH;
  size_t base = ((size_t)b * NN + n) * (3 * CC) + h * DD;
  float q0 = qkv[base + lane * 2], q1 = qkv[base + lane * 2 + 1];
  float k0 = qkv[base + CC + lane * 2], k1 = qkv[base + CC + lane * 2 + 1];
  float v0 = qkv[base + 2 * CC + lane * 2], v1 = qkv[base + 2 * CC + lane * 2 + 1];
  float sq = q0 * q0 + q1 * q1, sk = k0 * k0 + k1 * k1;
#pragma unroll
  for (int m = 16; m; m >>= 1) {
    sq += __shfl_xor(sq, m, 32);
    sk += __shfl_xor(sk, m, 32);
  }
  float rq = rsqrtf(sq * (1.f / DD) + 1e-6f);
  float rk = rsqrtf(sk * (1.f / DD) + 1e-6f);
  size_t o = ((size_t)bh * NN + n) * DD + lane * 2;
  qnb[o] = f2bf(q0 * rq); qnb[o + 1] = f2bf(q1 * rq);
  knb[o] = f2bf(k0 * rk); knb[o + 1] = f2bf(k1 * rk);
  vb[o]  = f2bf(v0);      vb[o + 1]  = f2bf(v1);
}

// ---------------- Windowed attention (WMMA, ~215KB LDS on 320KB WGP) ----------------
// block = (b,h,win,half): 96 queries x 256 keys, 6 waves, one 16-row q-tile each
#define QH 96
#define KLP 72    // k rows: 144B stride (36-bank step, gcd4 -> full-bank b128)
#define VLP 264   // v^T rows: 528B stride (== 4 banks mod 64)
#define SCP 257   // score rows: f32, stride == 1 bank mod 64 (softmax scans)
#define PRP 264   // prob rows
#define KL_OFF  0
#define VLT_OFF (WINSZ * KLP * 2)                       // 36864
#define SC_OFF  (VLT_OFF + DD * VLP * 2)                // 70656
#define PR_OFF  (SC_OFF + QH * SCP * 4)                 // 169344
#define ATTN_LDS (PR_OFF + QH * PRP * 2)                // 220032
__global__ __launch_bounds__(192) void attn_kernel(
    const bf16_t* __restrict__ qn, const bf16_t* __restrict__ kn,
    const bf16_t* __restrict__ vv, float* __restrict__ ctx) {
  extern __shared__ char smem[];
  bf16_t* kl  = (bf16_t*)(smem + KL_OFF);     // [256][KLP]
  bf16_t* vlt = (bf16_t*)(smem + VLT_OFF);    // [64][VLP]  (transposed V)
  float*  sc  = (float*)(smem + SC_OFF);      // [96][SCP]
  bf16_t* pr  = (bf16_t*)(smem + PR_OFF);     // [96][PRP]

  int bid = blockIdx.x;
  int halfb = bid & 1;
  int win = (bid >> 1) & (NWIN - 1);
  int h = (bid >> 5) & (HH - 1);
  int b = bid >> 9;
  int start = win * STRIDEW;
  int tid = threadIdx.x, lane = tid & 31, wv = tid >> 5;
  int laneM = lane & 15, half = lane >> 4;
  const int kb0 = half ? 8 : 0, kbB = half ? 16 : 0;
  size_t hbase = ((size_t)(b * HH + h)) * NN * DD;

  // stage K (async, contiguous dest) and V (transposed scatter)
  for (int c = tid; c < (WINSZ * DD) / 8; c += 192) {
    int j = c >> 3, d0 = (c & 7) * 8;
    int kj = start + j; if (kj >= NN) kj = NN - 1;
    stage16(kn + hbase + (size_t)kj * DD + d0, &kl[j * KLP + d0]);
    union { uint4 u; bf16_t e[8]; } tv;
    tv.u = *(const uint4*)(vv + hbase + (size_t)kj * DD + d0);
#pragma unroll
    for (int i = 0; i < 8; ++i) vlt[(d0 + i) * VLP + j] = tv.e[i];
  }
  stage_wait();
  __syncthreads();

  const v8f vzero = {0.f, 0.f, 0.f, 0.f, 0.f, 0.f, 0.f, 0.f};
  int qrow0 = start + halfb * QH + wv * 16;

  // scores = (qn . kn^T) * D^-0.5  -> LDS
  for (int kt = 0; kt < 16; ++kt) {
    v8f c = vzero;
#pragma unroll
    for (int kk = 0; kk < DD; kk += 32) {
      v16bf a = load_fragA(qn + hbase + (size_t)(qrow0 + laneM) * DD + kk, kb0);
      v16bf bb = load_frag16(&kl[(kt * 16 + laneM) * KLP + kk + kbB]);
      c = wmma_bf16(a, bb, c);
    }
#pragma unroll
    for (int r = 0; r < 8; ++r) {
      int m = wv * 16 + r + half * 8;
      sc[m * SCP + kt * 16 + laneM] = c[r] * 0.125f;
    }
  }
  __syncthreads();

  // in-LDS softmax with validity mask, repack probs as bf16
  if (tid < QH) {
    float* row = sc + tid * SCP;
    int nvalid = NN - start; if (nvalid > WINSZ) nvalid = WINSZ;
    float mx = -3.0e38f;
    for (int j = 0; j < nvalid; ++j) mx = fmaxf(mx, row[j]);
    float s = 0.f;
    for (int j = 0; j < nvalid; ++j) { float e = __expf(row[j] - mx); row[j] = e; s += e; }
    float inv = 1.f / s;
    bf16_t* prow = pr + tid * PRP;
    for (int j = 0; j < nvalid; ++j) prow[j] = f2bf(row[j] * inv);
    for (int j = nvalid; j < WINSZ; ++j) prow[j] = f2bf(0.f);
  }
  __syncthreads();

  // out = probs(96x256) . v(256x64), V read transposed -> contiguous B frags
  v8f acc[4] = {vzero, vzero, vzero, vzero};
  for (int kt = 0; kt < WINSZ / 32; ++kt) {
    v16bf a = load_fragA(pr + (wv * 16 + laneM) * PRP + kt * 32, kb0);
#pragma unroll
    for (int nt = 0; nt < 4; ++nt) {
      v16bf bb = load_frag16(&vlt[(nt * 16 + laneM) * VLP + kt * 32 + kbB]);
      acc[nt] = wmma_bf16(a, bb, acc[nt]);
    }
  }
#pragma unroll
  for (int nt = 0; nt < 4; ++nt) {
    int d = nt * 16 + laneM;
#pragma unroll
    for (int r = 0; r < 8; ++r) {
      int n = qrow0 + r + half * 8;
      ctx[(((size_t)b * NN + n) * HH + h) * DD + d] = acc[nt][r];
    }
  }
}

// ---------------- FIR conv (3/7 tap) + cross-head mix ----------------
__global__ __launch_bounds__(256) void fir_mix_kernel(
    const float* __restrict__ qkv, const float* __restrict__ fir_s,
    const float* __restrict__ fir_l, const float* __restrict__ mix,
    float* __restrict__ ps, float* __restrict__ pl) {
  __shared__ float ys[CC], yl[CC];
  int bn = blockIdx.x;
  int n = bn % NN, b = bn / NN;
  for (int i = threadIdx.x; i < CC; i += 256) {
    size_t vcol = 2 * CC + i;
    float a = 0.f;
#pragma unroll
    for (int t = 0; t < 3; ++t) {
      int np = n - 2 + t;
      if (np >= 0) a += fir_s[i * 3 + t] * qkv[((size_t)b * NN + np) * (3 * CC) + vcol];
    }
    ys[i] = a;
    float c = 0.f;
#pragma unroll
    for (int t = 0; t < 7; ++t) {
      int np = n - 6 + t;
      if (np >= 0) c += fir_l[i * 7 + t] * qkv[((size_t)b * NN + np) * (3 * CC) + vcol];
    }
    yl[i] = c;
  }
  __syncthreads();
  for (int i = threadIdx.x; i < CC; i += 256) {
    int h = i >> 6, d = i & 63;
    float as = ys[i], al = yl[i];
#pragma unroll
    for (int g = 0; g < HH; ++g) {
      float m = mix[h * HH + g];
      as += m * ys[g * DD + d];
      al += m * yl[g * DD + d];
    }
    size_t o = (size_t)bn * CC + i;
    ps[o] = as; pl[o] = al;
  }
}

// ---------------- Delta rule: sequential scan, S columns in registers ----------------
__global__ __launch_bounds__(64) void delta_kernel(
    const float* __restrict__ qkv, float* __restrict__ dout) {
  __shared__ float kk[DD], qq[DD];
  int bh = blockIdx.x;
  int h = bh % HH, b = bh / HH;
  int j = threadIdx.x;           // owns S[:, j]
  float S[DD];
#pragma unroll
  for (int d = 0; d < DD; ++d) S[d] = 0.f;
  for (int n = 0; n < NN; ++n) {
    size_t base = ((size_t)b * NN + n) * (3 * CC) + h * DD;
    kk[j] = qkv[base + CC + j];
    qq[j] = qkv[base + j];
    float v_j = qkv[base + 2 * CC + j];
    __syncthreads();
    float pred = 0.f;
#pragma unroll
    for (int d = 0; d < DD; ++d) pred += kk[d] * S[d];
    float beta = 1.0f - 0.9f * (float)n / (float)(NN - 1);
    float err = beta * (v_j - pred);
    float o = 0.f;
#pragma unroll
    for (int d = 0; d < DD; ++d) { S[d] += kk[d] * err; o += qq[d] * S[d]; }
    dout[((size_t)bh * NN + n) * DD + j] = o;
    __syncthreads();
  }
}

// ---------------- Fused gating epilogue + entropies ----------------
__global__ __launch_bounds__(256) void fuse_kernel(
    const float* __restrict__ x, const float* __restrict__ qkv,
    const float* __restrict__ hdnraw, const float* __restrict__ ps,
    const float* __restrict__ pl, const float* __restrict__ delta,
    const float* __restrict__ identity, const float* __restrict__ ctx,
    const float* __restrict__ w_gate, const float* __restrict__ b_gate,
    const float* __restrict__ w_prune, const float* __restrict__ b_prune,
    const float* __restrict__ w_f2, const float* __restrict__ b_f2,
    const float* __restrict__ id_static_logit, const float* __restrict__ b_f1,
    bf16_t* __restrict__ fusedb, float* __restrict__ ent) {
  __shared__ float xr[CC], hr[CC];
  __shared__ float dg[HH], pw[PP], gw[HH * PP], gs[HH * PP];
  int bn = blockIdx.x;
  int n = bn % NN, b = bn / NN;
  size_t rbase = (size_t)bn * CC;
  for (int i = threadIdx.x; i < CC; i += 256) {
    xr[i] = x[rbase + i];
    float hv = hdnraw[rbase + i] + b_f1[i];
    hr[i] = 0.5f * hv * (1.f + erff(hv * 0.70710678f));   // exact gelu
  }
  __syncthreads();
  int t = threadIdx.x;
  if (t < HH) {
    float s = b_gate[t];
    for (int i = 0; i < CC; ++i) s += xr[i] * w_gate[i * HH + t];
    dg[t] = 1.f / (1.f + __expf(-s));
  } else if (t < HH + PP) {
    int p = t - HH;
    float s = b_prune[p];
    for (int i = 0; i < CC; ++i) s += xr[i] * w_prune[i * PP + p];
    pw[p] = 1.f / (1.f + __expf(-s));
  } else if (t >= 32 && t < 32 + HH * PP) {
    int jg = t - 32;
    float s = b_f2[jg];
    for (int i = 0; i < CC; ++i) s += hr[i] * w_f2[i * (HH * PP) + jg];
    gw[jg] = s;
  }
  __syncthreads();
  if (t < HH) {
    float mx = -3e38f;
    for (int p = 0; p < PP; ++p) mx = fmaxf(mx, gw[t * PP + p]);
    float s = 0.f, e[PP];
    for (int p = 0; p < PP; ++p) { e[p] = __expf(gw[t * PP + p] - mx); s += e[p]; }
    float inv = 1.f / s, ge = 0.f;
    for (int p = 0; p < PP; ++p) {
      float g = e[p] * inv;
      gs[t * PP + p] = g;
      ge -= g * __logf(g + 1e-8f);
    }
    atomicAdd(&ent[0], ge);
  } else if (t < HH + PP) {
    int p = t - HH;
    float v = fminf(fmaxf(pw[p], 1e-6f), 1.f - 1e-6f);
    atomicAdd(&ent[1], -(v * __logf(v) + (1.f - v) * log1pf(-v)));
  }
  __syncthreads();
  for (int i = t; i < CC; i += 256) {
    int h = i >> 6, d = i & 63;
    float w0 = gs[h * PP + 0] * pw[0], w1 = gs[h * PP + 1] * pw[1],
          w2 = gs[h * PP + 2] * pw[2], w3 = gs[h * PP + 3] * pw[3],
          w4 = gs[h * PP + 4] * pw[4];
    float vh = qkv[(size_t)bn * (3 * CC) + 2 * CC + i];
    float sg = 1.f / (1.f + __expf(-id_static_logit[h]));
    float idv = identity[rbase + i] * dg[h] * sg;
    size_t dlt = (((size_t)(b * HH + h)) * NN + n) * DD + d;
    float f = w0 * ps[rbase + i] + w1 * pl[rbase + i] + w2 * delta[dlt] +
              w3 * vh + w4 * idv + ctx[rbase + i];
    fusedb[rbase + i] = f2bf(f);
  }
}

__global__ void zero_ent_kernel(float* ent) { ent[0] = 0.f; ent[1] = 0.f; }
__global__ void finalize_ent_kernel(const float* ent, float* out) {
  out[(size_t)BB * NN * CC] =
      ent[0] / (float)(BB * NN * HH) + ent[1] / (float)(BB * NN * PP);
}

// ---------------- Host orchestration ----------------
extern "C" void kernel_launch(void* const* d_in, const int* in_sizes, int n_in,
                              void* d_out, int out_size, void* d_ws, size_t ws_size,
                              hipStream_t stream) {
  const float* x       = (const float*)d_in[0];
  const float* w_qkv   = (const float*)d_in[1];
  const float* w_proj  = (const float*)d_in[2];
  const float* b_proj  = (const float*)d_in[3];
  const float* fir_s   = (const float*)d_in[4];
  const float* fir_l   = (const float*)d_in[5];
  const float* mix     = (const float*)d_in[6];
  const float* w_id    = (const float*)d_in[7];
  const float* id_sl   = (const float*)d_in[8];
  const float* w_gate  = (const float*)d_in[9];
  const float* b_gate  = (const float*)d_in[10];
  const float* w_prune = (const float*)d_in[11];
  const float* b_prune = (const float*)d_in[12];
  const float* w_f1    = (const float*)d_in[13];
  const float* b_f1    = (const float*)d_in[14];
  const float* w_f2    = (const float*)d_in[15];
  const float* b_f2    = (const float*)d_in[16];
  float* out = (float*)d_out;

  char* ws = (char*)d_ws;
  auto alloc = [&](size_t bytes) {
    char* p = ws;
    ws += (bytes + 255) & ~(size_t)255;
    return p;
  };
  const size_t MROWS = (size_t)BB * NN;                 // 6144
  bf16_t* xb     = (bf16_t*)alloc(MROWS * CC * 2);
  bf16_t* wqkvb  = (bf16_t*)alloc((size_t)CC * 3 * CC * 2);
  bf16_t* widb   = (bf16_t*)alloc((size_t)CC * CC * 2);
  bf16_t* wf1b   = (bf16_t*)alloc((size_t)CC * CC * 2);
  bf16_t* wprojb = (bf16_t*)alloc((size_t)CC * CC * 2);
  bf16_t* fusedb = (bf16_t*)alloc(MROWS * CC * 2);
  bf16_t* qnb    = (bf16_t*)alloc(MROWS * CC * 2);
  bf16_t* knb    = (bf16_t*)alloc(MROWS * CC * 2);
  bf16_t* vb     = (bf16_t*)alloc(MROWS * CC * 2);
  float* qkv     = (float*)alloc(MROWS * 3 * CC * 4);
  float* ctx     = (float*)alloc(MROWS * CC * 4);
  float* psb     = (float*)alloc(MROWS * CC * 4);
  float* plb     = (float*)alloc(MROWS * CC * 4);
  float* dlt     = (float*)alloc(MROWS * CC * 4);
  float* idb     = (float*)alloc(MROWS * CC * 4);
  float* hdnraw  = (float*)alloc(MROWS * CC * 4);
  float* ent     = (float*)alloc(256);

  zero_ent_kernel<<<1, 1, 0, stream>>>(ent);

  auto cvt = [&](const float* s, bf16_t* d, size_t n) {
    cvt_bf16_kernel<<<(unsigned)((n + 255) / 256), 256, 0, stream>>>(s, d, n);
  };
  cvt(x, xb, MROWS * CC);
  cvt(w_qkv, wqkvb, (size_t)CC * 3 * CC);
  cvt(w_id, widb, (size_t)CC * CC);
  cvt(w_f1, wf1b, (size_t)CC * CC);
  cvt(w_proj, wprojb, (size_t)CC * CC);

  auto gemm = [&](const bf16_t* A, const bf16_t* Bm, float* Cm,
                  const float* bias, int M, int Nn, int K) {
    dim3 grid(Nn / BN, M / BM);
    gemm_bf16_kernel<<<grid, 256, 0, stream>>>(A, Bm, Cm, bias, M, Nn, K);
  };
  gemm(xb, wqkvb, qkv, nullptr, (int)MROWS, 3 * CC, CC);
  gemm(xb, widb, idb, nullptr, (int)MROWS, CC, CC);
  gemm(xb, wf1b, hdnraw, nullptr, (int)MROWS, CC, CC);

  qkv_post_kernel<<<(BB * HH * NN) / 8, 256, 0, stream>>>(qkv, qnb, knb, vb);

  (void)hipFuncSetAttribute((const void*)attn_kernel,
                            hipFuncAttributeMaxDynamicSharedMemorySize, ATTN_LDS);
  attn_kernel<<<BB * HH * NWIN * 2, 192, ATTN_LDS, stream>>>(qnb, knb, vb, ctx);

  fir_mix_kernel<<<(unsigned)MROWS, 256, 0, stream>>>(qkv, fir_s, fir_l, mix, psb, plb);
  delta_kernel<<<BB * HH, 64, 0, stream>>>(qkv, dlt);

  fuse_kernel<<<(unsigned)MROWS, 256, 0, stream>>>(
      x, qkv, hdnraw, psb, plb, dlt, idb, ctx, w_gate, b_gate, w_prune, b_prune,
      w_f2, b_f2, id_sl, b_f1, fusedb, ent);

  gemm(fusedb, wprojb, out, b_proj, (int)MROWS, CC, CC);
  finalize_ent_kernel<<<1, 1, 0, stream>>>(ent, out);
}
